// MultiHeadAttention_67697274520364
// MI455X (gfx1250) — matrix operand order repaired and checked by reading, not compiled, not run on hardware
//
#include <hip/hip_runtime.h>

#define SEQ 240
#define DM  4096
#define NH  16
#define HD  256

typedef __attribute__((ext_vector_type(16))) __bf16 bf16x16;
typedef __attribute__((ext_vector_type(8)))  float  f32x8;
typedef __attribute__((ext_vector_type(4)))  float  f32x4;

union FragAB { bf16x16 v; uint4 q[2]; };

#if defined(__has_builtin)
#if __has_builtin(__builtin_amdgcn_cvt_pk_bf16_f32)
#define HAVE_CVT_PK_BF16 1
#endif
#endif

// pack two fp32 -> two bf16 (round-to-nearest, ties-away): 2x v_add + v_perm_b32
__device__ __forceinline__ unsigned pk2(float lo, float hi) {
#ifdef HAVE_CVT_PK_BF16
  auto p = __builtin_amdgcn_cvt_pk_bf16_f32(lo, hi);   // v_cvt_pk_bf16_f32
  return __builtin_bit_cast(unsigned, p);
#else
  const unsigned ul = __float_as_uint(lo) + 0x8000u;
  const unsigned uh = __float_as_uint(hi) + 0x8000u;
  // D.b0=S1.b2 D.b1=S1.b3 D.b2=S0.b2 D.b3=S0.b3 -> {hi[31:16], lo[31:16]}
  return __builtin_amdgcn_perm(uh, ul, 0x07060302u);
#endif
}

__device__ __forceinline__ unsigned short f2bf(float f) {
  return (unsigned short)((__float_as_uint(f) + 0x8000u) >> 16);
}

__device__ __forceinline__ f32x8 zero8() {
  f32x8 z;
#pragma unroll
  for (int i = 0; i < 8; ++i) z[i] = 0.0f;
  return z;
}

// single-use streams: TH=NT load, don't pollute L2 (keep it for activations)
__device__ __forceinline__ f32x4 ldnt4(const float* p) {
  return __builtin_nontemporal_load(reinterpret_cast<const f32x4*>(p));
}

// CDNA5 async VMEM->LDS copy (tracked on ASYNCcnt). Inline asm: portable
// across ROCm7.2/amdgpu-toolchain (the TDM builtin arity is not).
__device__ __forceinline__ void async_cp16(unsigned lds_byte, const void* g) {
  asm volatile("global_load_async_to_lds_b128 %0, %1, off"
               :: "v"(lds_byte), "v"(g) : "memory");
}
__device__ __forceinline__ void wait_async0() {
  asm volatile("s_wait_asynccnt 0x0" ::: "memory");
}

// ---------------------------------------------------------------------------
// Kernel 0: fp32 -> bf16 activation conversion (done ONCE; removes all
// per-tile activation conversion from the GEMM).
// ---------------------------------------------------------------------------
__global__ __launch_bounds__(256) void cvt_bf16(const float* __restrict__ src,
                                                unsigned short* __restrict__ dst,
                                                int n4) {
  const int i = blockIdx.x * 256 + threadIdx.x;
  if (i < n4) {
    f32x4 f = ldnt4(src + (size_t)i * 4);
    reinterpret_cast<uint2*>(dst)[i] = make_uint2(pk2(f[0], f[1]), pk2(f[2], f[3]));
  }
}

// ---------------------------------------------------------------------------
// Kernel 1: out = X @ W^T + b (X bf16, W fp32 streamed ONCE non-temporally,
// bf16 out). Workgroup covers full M=240 for a 64-wide N slice => every
// weight element loaded+converted exactly once. 8 waves: wave w owns M-tiles
// {2w,2w+1}, all 4 N-tiles. Double-buffered LDS weight staging, one barrier
// per K-step. grid = (64 N-blocks, 3 projections). V stored transposed.
// ---------------------------------------------------------------------------
__global__ __launch_bounds__(256) void qkv_gemm(
    const unsigned short* __restrict__ X1, const unsigned short* __restrict__ X2,
    const float* __restrict__ Wq, const float* __restrict__ Wk, const float* __restrict__ Wv,
    const float* __restrict__ bq, const float* __restrict__ bk, const float* __restrict__ bv,
    unsigned short* __restrict__ outQ, unsigned short* __restrict__ outK,
    unsigned short* __restrict__ outVt) {
  __shared__ unsigned short lsB[2][64 * 40];   // stride 40: bank-conflict free

  const unsigned short* X; const float* W; const float* bias;
  unsigned short* out; int tr;
  if (blockIdx.y == 0)      { X = X1; W = Wq; bias = bq; out = outQ;  tr = 0; }
  else if (blockIdx.y == 1) { X = X2; W = Wk; bias = bk; out = outK;  tr = 0; }
  else                      { X = X2; W = Wv; bias = bv; out = outVt; tr = 1; }

  const int tid  = threadIdx.x;
  const int wave = tid >> 5, lane = tid & 31;
  const int half = lane >> 4, mloc = lane & 15;
  const int n0   = blockIdx.x * 64;
  const int srow = tid >> 2;     // 0..63 staging row
  const int sch  = tid & 3;      // 0..3  (8-float chunk)
  const float* wsrc = W + (size_t)(n0 + srow) * DM + sch * 8;
  unsigned short* sdst0 = &lsB[0][srow * 40 + sch * 8];
  unsigned short* sdst1 = &lsB[1][srow * 40 + sch * 8];

  f32x8 acc[2][4];
#pragma unroll
  for (int mi = 0; mi < 2; ++mi)
#pragma unroll
    for (int j = 0; j < 4; ++j) acc[mi][j] = zero8();

  // prologue: stage tile 0
  {
    f32x4 w0 = ldnt4(wsrc);
    f32x4 w1 = ldnt4(wsrc + 4);
    *reinterpret_cast<uint4*>(sdst0) =
        make_uint4(pk2(w0[0], w0[1]), pk2(w0[2], w0[3]), pk2(w1[0], w1[1]), pk2(w1[2], w1[3]));
  }
  __syncthreads();

  for (int kb = 0; kb < DM / 32; ++kb) {
    const int cur = kb & 1;
    const int k0  = kb * 32;
    const bool hasNext = (kb + 1) < DM / 32;
    f32x4 w0, w1;
    if (hasNext) {  // issue next tile's global loads before compute
      w0 = ldnt4(wsrc + k0 + 32);
      w1 = ldnt4(wsrc + k0 + 36);
    }
    // A fragments straight from global bf16 (L2-resident, k-contiguous 16B)
    FragAB a[2];
#pragma unroll
    for (int mi = 0; mi < 2; ++mi) {
      const unsigned short* px = X + (size_t)((2 * wave + mi) * 16 + mloc) * DM + k0;
      a[mi].q[0] = *reinterpret_cast<const uint4*>(px + half * 8);
      a[mi].q[1] = *reinterpret_cast<const uint4*>(px + 16 + half * 8);
    }
#pragma unroll
    for (int j = 0; j < 4; ++j) {
      FragAB b;
      const unsigned short* pb = &lsB[cur][(j * 16 + mloc) * 40 + half * 8];
      b.q[0] = *reinterpret_cast<const uint4*>(pb);
      b.q[1] = *reinterpret_cast<const uint4*>(pb + 16);
#pragma unroll
      for (int mi = 0; mi < 2; ++mi)
        acc[mi][j] = __builtin_amdgcn_wmma_f32_16x16x32_bf16(false, a[mi].v, false, b.v,
                                                             (short)0, acc[mi][j], false, false);
    }
    if (hasNext) {  // convert + store into the other buffer
      unsigned short* sd = (cur == 0) ? sdst1 : sdst0;
      *reinterpret_cast<uint4*>(sd) =
          make_uint4(pk2(w0[0], w0[1]), pk2(w0[2], w0[3]), pk2(w1[0], w1[1]), pk2(w1[2], w1[3]));
    }
    __syncthreads();
  }

  // epilogue: bias + bf16 store (wave 7's second tile is out of range -> skipped)
#pragma unroll
  for (int mi = 0; mi < 2; ++mi) {
    const int mt = 2 * wave + mi;
    if (mt > 14) continue;
#pragma unroll
    for (int j = 0; j < 4; ++j) {
      const int gn = n0 + j * 16 + mloc;
      const float bn = bias[gn];
#pragma unroll
      for (int r = 0; r < 8; ++r) {
        const int gm = mt * 16 + r + half * 8;
        const unsigned short v = f2bf(acc[mi][j][r] + bn);
        if (!tr) out[(size_t)gm * DM + gn] = v;
        else     out[(size_t)gn * 256 + gm] = v;   // Vt[d][q], stride 256
      }
    }
  }
}

// ---------------------------------------------------------------------------
// Kernel 2: scores = (Q @ K^T)/16, softmax over kk, store W^T as Wt[h][kk][q].
// K slice staged into LDS with global_load_async_to_lds_b128 (double-buffered,
// ASYNCcnt + barrier). One wave per (head, q-tile); 120 accumulator VGPRs.
// ---------------------------------------------------------------------------
__global__ __launch_bounds__(128) void attn_scores_softmax(
    const unsigned short* __restrict__ Q,
    const unsigned short* __restrict__ K,
    unsigned short* __restrict__ Wt) {
  __shared__ unsigned short lsK[2][240 * 40];

  const int tid  = threadIdx.x;
  const int wave = tid >> 5, lane = tid & 31;
  const int half = lane >> 4, mloc = lane & 15;
  const int h = blockIdx.y;
  int qt = blockIdx.x * 4 + wave;
  if (qt > 14) qt = 14;   // dead wave duplicates qt=14 (same bytes) — keeps
                          // barrier membership and EXEC=all-1s for WMMA

  auto stage = [&](int kb, int buf) {
    const int k0 = kb * 32;
    for (int u = tid; u < 960; u += 128) {          // 240 rows x 4 chunks
      const int row = u >> 2, ch = u & 3;
      const unsigned lds = (unsigned)(uintptr_t)&lsK[buf][row * 40 + ch * 8];
      async_cp16(lds, K + (size_t)row * DM + h * HD + k0 + ch * 8);
    }
  };

  f32x8 acc[15];
#pragma unroll
  for (int j = 0; j < 15; ++j) acc[j] = zero8();

  stage(0, 0);
  wait_async0();
  __syncthreads();

  const size_t qbase = (size_t)(qt * 16 + mloc) * DM + h * HD;
  for (int kb = 0; kb < 8; ++kb) {
    const int cur = kb & 1;
    if (kb + 1 < 8) stage(kb + 1, cur ^ 1);
    FragAB a;
    a.q[0] = *reinterpret_cast<const uint4*>(Q + qbase + kb * 32 + half * 8);
    a.q[1] = *reinterpret_cast<const uint4*>(Q + qbase + kb * 32 + 16 + half * 8);
#pragma unroll
    for (int j = 0; j < 15; ++j) {
      FragAB b;
      const unsigned short* pb = &lsK[cur][(j * 16 + mloc) * 40 + half * 8];
      b.q[0] = *reinterpret_cast<const uint4*>(pb);
      b.q[1] = *reinterpret_cast<const uint4*>(pb + 16);
      acc[j] = __builtin_amdgcn_wmma_f32_16x16x32_bf16(false, a.v, false, b.v,
                                                       (short)0, acc[j], false, false);
    }
    wait_async0();
    __syncthreads();
  }

  // row softmax (rows live in 16-lane halves; xor-shuffles stay in-half)
  const float sc = 0.0625f;  // 1/sqrt(256)
#pragma unroll
  for (int r = 0; r < 8; ++r) {
    float mx = -1e30f;
#pragma unroll
    for (int j = 0; j < 15; ++j) { acc[j][r] *= sc; mx = fmaxf(mx, acc[j][r]); }
#pragma unroll
    for (int off = 1; off < 16; off <<= 1) mx = fmaxf(mx, __shfl_xor(mx, off, 32));
    float sum = 0.0f;
#pragma unroll
    for (int j = 0; j < 15; ++j) {
      const float e = __expf(acc[j][r] - mx);
      acc[j][r] = e;
      sum += e;
    }
#pragma unroll
    for (int off = 1; off < 16; off <<= 1) sum += __shfl_xor(sum, off, 32);
    const float inv = 1.0f / sum;

    const int q = qt * 16 + r + half * 8;
#pragma unroll
    for (int j = 0; j < 15; ++j) {
      const int kk = j * 16 + mloc;
      Wt[(size_t)(h * SEQ + kk) * 256 + q] = f2bf(acc[j][r] * inv);
    }
  }
}

// ---------------------------------------------------------------------------
// Kernel 3: out[kk, h*256+d] = sum_q Wt[h][kk][q] * Vt[h*256+d][q].
// K = q = 240 padded to 8x32 with register zeros (pad region is poison).
// ---------------------------------------------------------------------------
__global__ __launch_bounds__(128) void attn_out(
    const unsigned short* __restrict__ Wt,
    const unsigned short* __restrict__ Vt,
    float* __restrict__ out) {
  const int tid  = threadIdx.x;
  const int wave = tid >> 5;   // d range wave*64
  const int lane = tid & 31;
  const int half = lane >> 4, mloc = lane & 15;
  const int h  = blockIdx.y;
  const int mt = blockIdx.x;   // kk tile 0..14

  f32x8 acc[4];
#pragma unroll
  for (int j = 0; j < 4; ++j) acc[j] = zero8();

  const unsigned short* wrow = Wt + (size_t)(h * SEQ + mt * 16 + mloc) * 256;
  const uint4 zq = make_uint4(0u, 0u, 0u, 0u);

  for (int kb = 0; kb < 8; ++kb) {
    FragAB a;
    const int c0 = kb * 32 + half * 8;         // always < 240
    const int c1 = kb * 32 + 16 + half * 8;    // >= 240 on last step
    a.q[0] = *reinterpret_cast<const uint4*>(wrow + c0);
    a.q[1] = (c1 < SEQ) ? *reinterpret_cast<const uint4*>(wrow + c1) : zq;
#pragma unroll
    for (int j = 0; j < 4; ++j) {
      FragAB b;
      const int d = h * HD + wave * 64 + j * 16 + mloc;
      const int koff = kb * 32 + half * 8;
      if (koff < SEQ) {
        const unsigned short* pv = Vt + (size_t)d * 256 + koff;
        b.q[0] = *reinterpret_cast<const uint4*>(pv);
        b.q[1] = (koff + 16 < SEQ) ? *reinterpret_cast<const uint4*>(pv + 16) : zq;
      } else {
        b.q[0] = zq; b.q[1] = zq;
      }
      acc[j] = __builtin_amdgcn_wmma_f32_16x16x32_bf16(false, a.v, false, b.v,
                                                       (short)0, acc[j], false, false);
    }
  }

#pragma unroll
  for (int j = 0; j < 4; ++j) {
    const int col = h * HD + wave * 64 + j * 16 + mloc;
#pragma unroll
    for (int r = 0; r < 8; ++r) {
      const int row = mt * 16 + r + half * 8;  // kk
      out[(size_t)row * DM + col] = acc[j][r];
    }
  }
}

// ---------------------------------------------------------------------------
extern "C" void kernel_launch(void* const* d_in, const int* in_sizes, int n_in,
                              void* d_out, int out_size, void* d_ws, size_t ws_size,
                              hipStream_t stream) {
  const float* input1 = (const float*)d_in[0];
  const float* input2 = (const float*)d_in[1];
  const float* Wq = (const float*)d_in[2];
  const float* bq = (const float*)d_in[3];
  const float* Wk = (const float*)d_in[4];
  const float* bk = (const float*)d_in[5];
  const float* Wv = (const float*)d_in[6];
  const float* bv = (const float*)d_in[7];
  float* out = (float*)d_out;

  // bf16 workspace (u16 units):
  // X1[240*4096] | X2[240*4096] | Q[240*4096] | K[240*4096] | Vt[4096*256] | Wt[3840*256]
  unsigned short* x1  = (unsigned short*)d_ws;
  unsigned short* x2  = x1  + (size_t)SEQ * DM;
  unsigned short* qws = x2  + (size_t)SEQ * DM;
  unsigned short* kws = qws + (size_t)SEQ * DM;
  unsigned short* vt  = kws + (size_t)SEQ * DM;
  unsigned short* wt  = vt  + (size_t)DM * 256;

  const int n4 = SEQ * DM / 4;
  cvt_bf16<<<dim3((n4 + 255) / 256), 256, 0, stream>>>(input1, x1, n4);
  cvt_bf16<<<dim3((n4 + 255) / 256), 256, 0, stream>>>(input2, x2, n4);

  qkv_gemm<<<dim3(DM / 64, 3), 256, 0, stream>>>(x1, x2, Wq, Wk, Wv, bq, bk, bv,
                                                 qws, kws, vt);

  attn_scores_softmax<<<dim3(4, NH), 128, 0, stream>>>(qws, kws, wt);
  attn_out<<<dim3(15, NH), 128, 0, stream>>>(wt, vt, out);
}
